// NNConvModel_25048249270792
// MI455X (gfx1250) — compile-verified, matrix operand order checked by reading, not run on hardware
//
#include <hip/hip_runtime.h>

// ---------------------------------------------------------------------------
// MI455X (gfx1250) implementation of the NNConv GNN forward pass.
// All linear layers run as f16-in/f32-accumulate GEMMs on
// v_wmma_f32_16x16x32_f16. The dominant cost (nn_lin2 -> einsum ->
// segment_sum, ~156 GFLOP across layers) is fused so the [E, nf*nout]
// per-edge weight tensor (1 GiB f32) is never materialized: each 16x16 block
// of w lives only in WMMA accumulators, is scaled by x[src] and scatter-added
// with f32 atomics. The fused kernel processes two edge tiles per wave so
// every B (weight) fragment feeds two WMMAs (32 flops/byte from L2), and the
// generic GEMM computes 16x32 tiles so every A (activation) fragment feeds
// two WMMAs (halves HBM-side activation reads).
// ---------------------------------------------------------------------------

typedef _Float16 h8  __attribute__((ext_vector_type(8)));
typedef _Float16 h16 __attribute__((ext_vector_type(16)));
typedef float    f8  __attribute__((ext_vector_type(8)));

#define LEAK_F 0.1f
#define BN_EPS 1e-5f

__device__ __forceinline__ void atom_add_f32(float* p, float v) {
#if __has_builtin(__builtin_amdgcn_flat_atomic_fadd_f32)
  __builtin_amdgcn_flat_atomic_fadd_f32(p, v);
#else
  atomicAdd(p, v);
#endif
}

// ---------------- weight packing: f32 [N,K] -> f16 [N,Kp], zero-padded -----
__global__ void pack_wt(const float* __restrict__ W, _Float16* __restrict__ Wh,
                        int N, int K, int Kp) {
  int idx = blockIdx.x * blockDim.x + threadIdx.x;
  int tot = N * Kp;
  if (idx >= tot) return;
  int n = idx / Kp, k = idx - n * Kp;
  Wh[idx] = (k < K) ? (_Float16)W[n * K + k] : (_Float16)0.f;
}

// f32 [K,N] -> f16 [Kp,N] (row-padded, for non-transposed B like `root`) ----
__global__ void pack_bk(const float* __restrict__ B, _Float16* __restrict__ Bh,
                        int K, int N, int Kp) {
  int idx = blockIdx.x * blockDim.x + threadIdx.x;
  int tot = Kp * N;
  if (idx >= tot) return;
  int k = idx / N, n = idx - k * N;
  Bh[idx] = (k < K) ? (_Float16)B[k * N + n] : (_Float16)0.f;
}

// ---------------- batchnorm statistics (f32 input) -------------------------
__global__ void bn_stats_f(const float* __restrict__ X, int M, int D,
                           float* __restrict__ mean, float* __restrict__ rstd) {
  int d = blockIdx.x;
  __shared__ float ss[256], sq[256];
  float s = 0.f, q = 0.f;
  for (int m = threadIdx.x; m < M; m += blockDim.x) {
    float v = X[m * D + d];
    s += v; q += v * v;
  }
  ss[threadIdx.x] = s; sq[threadIdx.x] = q;
  __syncthreads();
  for (int w = 128; w > 0; w >>= 1) {
    if (threadIdx.x < w) {
      ss[threadIdx.x] += ss[threadIdx.x + w];
      sq[threadIdx.x] += sq[threadIdx.x + w];
    }
    __syncthreads();
  }
  if (threadIdx.x == 0) {
    float mu = ss[0] / (float)M;
    float var = sq[0] / (float)M - mu * mu;
    mean[d] = mu; rstd[d] = rsqrtf(var + BN_EPS);
  }
}

// normalize f32 input and emit padded f16 ----------------------------------
__global__ void bn_cvt(const float* __restrict__ X, int M, int D,
                       _Float16* __restrict__ Y, int Sy,
                       const float* __restrict__ mean, const float* __restrict__ rstd,
                       const float* __restrict__ g, const float* __restrict__ b) {
  int idx = blockIdx.x * blockDim.x + threadIdx.x;
  int tot = M * Sy;
  if (idx >= tot) return;
  int m = idx / Sy, c = idx - m * Sy;
  _Float16 o = (_Float16)0.f;
  if (c < D) {
    float v = (X[m * D + c] - mean[c]) * rstd[c] * g[c] + b[c];
    o = (_Float16)v;
  }
  Y[idx] = o;
}

// ---------------- batchnorm over f16 activation buffer ---------------------
__global__ void bn_stats_h(const _Float16* __restrict__ X, int M, int S,
                           float* __restrict__ mean, float* __restrict__ rstd) {
  int d = blockIdx.x;
  __shared__ float ss[256], sq[256];
  float s = 0.f, q = 0.f;
  for (int m = threadIdx.x; m < M; m += blockDim.x) {
    float v = (float)X[(size_t)m * S + d];
    s += v; q += v * v;
  }
  ss[threadIdx.x] = s; sq[threadIdx.x] = q;
  __syncthreads();
  for (int w = 128; w > 0; w >>= 1) {
    if (threadIdx.x < w) {
      ss[threadIdx.x] += ss[threadIdx.x + w];
      sq[threadIdx.x] += sq[threadIdx.x + w];
    }
    __syncthreads();
  }
  if (threadIdx.x == 0) {
    float mu = ss[0] / (float)M;
    float var = sq[0] / (float)M - mu * mu;
    mean[d] = mu; rstd[d] = rsqrtf(var + BN_EPS);
  }
}

__global__ void bn_apply_h(_Float16* __restrict__ X, int M, int S, int D,
                           const float* __restrict__ mean, const float* __restrict__ rstd,
                           const float* __restrict__ g, const float* __restrict__ b) {
  int idx = blockIdx.x * blockDim.x + threadIdx.x;
  int tot = M * D;
  if (idx >= tot) return;
  int m = idx / D, d = idx - m * D;
  float v = (float)X[(size_t)m * S + d];
  v = (v - mean[d]) * rstd[d] * g[d] + b[d];
  X[(size_t)m * S + d] = (_Float16)v;
}

// ---------------- gather + concat [x_src | x_dst | e] -> padded f16 --------
__global__ void gather_concat(const _Float16* __restrict__ X, int Sx, int nf,
                              const _Float16* __restrict__ E, int Se, int ef,
                              const int* __restrict__ src, const int* __restrict__ dst,
                              _Float16* __restrict__ H, int Sc, int nE) {
  int idx = blockIdx.x * blockDim.x + threadIdx.x;
  int tot = nE * Sc;
  if (idx >= tot) return;
  int e = idx / Sc, c = idx - e * Sc;
  _Float16 v = (_Float16)0.f;
  if (c < nf)               v = X[(size_t)src[e] * Sx + c];
  else if (c < 2 * nf)      v = X[(size_t)dst[e] * Sx + (c - nf)];
  else if (c < 2 * nf + ef) v = E[(size_t)e * Se + (c - 2 * nf)];
  H[idx] = v;
}

// ---------------- generic WMMA GEMM ----------------------------------------
// C[M,Npad] = A[M,K]f16 @ (bt ? B[N,Kp]^T : B[Kp,N]) + bias (+addh/+addf)
// K multiple of 32; A/B zero-padded. One wave computes a 16x32 tile: the A
// fragment is loaded once per k-step and reused by two B columns (halves
// HBM-side activation traffic). Ch pad columns are written as 0 so the buffer
// can be re-consumed as a K-padded A.
__global__ void __launch_bounds__(32)
wmma_gemm(const _Float16* __restrict__ A, int lda,
          const _Float16* __restrict__ B, int ldb, int bt,
          const float* __restrict__ bias,
          const _Float16* __restrict__ addh, int ldah,
          const float* __restrict__ addf, int ldaf,
          _Float16* __restrict__ Ch, int ldch,
          float* __restrict__ Cf, int ldcf,
          int N, int K, int do_lrelu) {
  const int lane = threadIdx.x;
  const int m0 = blockIdx.y << 4;
  const int n0 = blockIdx.x << 5;          // 2 x 16-column subtiles
  const int l15 = lane & 15;
  const int kb  = (lane >> 4) << 3;        // +8 K-offset for upper half-wave
  const int mA  = m0 + l15;
  const int nnA = n0 + l15;
  const int nnB = n0 + 16 + l15;
  const int ncA = (nnA < N) ? nnA : (N - 1);
  const int ncB = (nnB < N) ? nnB : (N - 1);

  f8 c0 = {}, c1 = {};
  if (bt) {
    const _Float16* ap  = A + (size_t)mA * lda;
    const _Float16* bp0 = B + (size_t)ncA * ldb;
    const _Float16* bp1 = B + (size_t)ncB * ldb;
    for (int k = 0; k < K; k += 32) {
      h16 a, b0, b1;
      *(h8*)&a      = *(const h8*)(ap + k + kb);
      ((h8*)&a)[1]  = *(const h8*)(ap + k + kb + 16);
      *(h8*)&b0     = *(const h8*)(bp0 + k + kb);
      ((h8*)&b0)[1] = *(const h8*)(bp0 + k + kb + 16);
      *(h8*)&b1     = *(const h8*)(bp1 + k + kb);
      ((h8*)&b1)[1] = *(const h8*)(bp1 + k + kb + 16);
      __builtin_prefetch(ap + k + 64, 0, 1);
      c0 = __builtin_amdgcn_wmma_f32_16x16x32_f16(false, a, false, b0, (short)0,
                                                  c0, false, false);
      c1 = __builtin_amdgcn_wmma_f32_16x16x32_f16(false, a, false, b1, (short)0,
                                                  c1, false, false);
    }
  } else {
    for (int k = 0; k < K; k += 32) {
      h16 a, b0, b1;
      const _Float16* ap = A + (size_t)mA * lda + k + kb;
      *(h8*)&a     = *(const h8*)(ap);
      ((h8*)&a)[1] = *(const h8*)(ap + 16);
#pragma unroll
      for (int j = 0; j < 8; ++j) {
        b0[j]     = B[(size_t)(k + kb + j) * ldb + ncA];
        b0[8 + j] = B[(size_t)(k + kb + 16 + j) * ldb + ncA];
        b1[j]     = B[(size_t)(k + kb + j) * ldb + ncB];
        b1[8 + j] = B[(size_t)(k + kb + 16 + j) * ldb + ncB];
      }
      c0 = __builtin_amdgcn_wmma_f32_16x16x32_f16(false, a, false, b0, (short)0,
                                                  c0, false, false);
      c1 = __builtin_amdgcn_wmma_f32_16x16x32_f16(false, a, false, b1, (short)0,
                                                  c1, false, false);
    }
  }

  const int mb = (lane >> 4) << 3;
#pragma unroll
  for (int s = 0; s < 2; ++s) {
    const int nn = n0 + (s << 4) + l15;
    const bool inN = (nn < N);
    const float bv = (bias && inN) ? bias[nn] : 0.f;
#pragma unroll
    for (int r = 0; r < 8; ++r) {
      const int mr = m0 + mb + r;
      float v = (s == 0 ? c0[r] : c1[r]) + bv;
      if (addh && inN) v += (float)addh[(size_t)mr * ldah + nn];
      if (addf && inN) v += addf[(size_t)mr * ldaf + nn];
      if (do_lrelu) v = (v >= 0.f) ? v : LEAK_F * v;
      if (Ch) Ch[(size_t)mr * ldch + nn] = inN ? (_Float16)v : (_Float16)0.f;
      if (Cf && inN) Cf[(size_t)mr * ldcf + nn] = v;
    }
  }
}

// ---------------- fused NNConv ---------------------------------------------
// Per wave: two 16-edge tiles (32 edges). H fragments (A) live in registers
// (KSN compile-time -> static register indexing, no scratch). Each W2
// fragment (B) feeds two WMMAs. Each 16x16 block of the per-edge weight
// tensor exists only in accumulators, is scaled by x[src] (LDS broadcast) and
// scatter-added into AGG[dst] with f32 atomics.
template <int KSN>
__global__ void __launch_bounds__(32)
nnconv_fused(const _Float16* __restrict__ WN,                  // H: [E, KW]
             const _Float16* __restrict__ W2,                  // [NF*NOUT, KW]
             const float* __restrict__ B2,                     // [NF*NOUT]
             const _Float16* __restrict__ X, int Sx, int NF, int nfsh,
             const int* __restrict__ src, const int* __restrict__ dst,
             float* __restrict__ AGG, int NOUT) {
  const int KW = KSN * 32;
  const int lane = threadIdx.x;
  const int e0 = blockIdx.x << 5;            // 32 edges
  __shared__ int sid[32], did[32];
  __shared__ _Float16 xs[32 * 64];
  sid[lane] = src[e0 + lane];
  did[lane] = dst[e0 + lane];
  __syncthreads();
  for (int idx = lane; idx < (32 << nfsh); idx += 32) {
    int r = idx >> nfsh, cix = idx & (NF - 1);
    xs[idx] = X[(size_t)sid[r] * Sx + cix];
  }
  __syncthreads();

  const int l15 = lane & 15;
  const int kb  = (lane >> 4) << 3;
  const int mb  = (lane >> 4) << 3;

  h16 afr0[KSN], afr1[KSN];
  {
    const _Float16* a0 = WN + (size_t)(e0 + l15) * KW + kb;
    const _Float16* a1 = WN + (size_t)(e0 + 16 + l15) * KW + kb;
#pragma unroll
    for (int ks = 0; ks < KSN; ++ks) {
      h16 a;
      *(h8*)&a     = *(const h8*)(a0 + ks * 32);
      ((h8*)&a)[1] = *(const h8*)(a0 + ks * 32 + 16);
      afr0[ks] = a;
      *(h8*)&a     = *(const h8*)(a1 + ks * 32);
      ((h8*)&a)[1] = *(const h8*)(a1 + ks * 32 + 16);
      afr1[ks] = a;
    }
  }

  for (int ot = 0; ot < NOUT; ot += 16) {
    f8 acc0 = {}, acc1 = {};
    for (int i = 0; i < NF; ++i) {
      const int wrow = i * NOUT + ot + l15;
      const _Float16* wp = W2 + (size_t)wrow * KW + kb;
      __builtin_prefetch(wp + (size_t)NOUT * KW, 0, 1);  // next i row
      f8 c0 = {}, c1 = {};
#pragma unroll
      for (int ks = 0; ks < KSN; ++ks) {
        h16 b;
        *(h8*)&b     = *(const h8*)(wp + ks * 32);
        ((h8*)&b)[1] = *(const h8*)(wp + ks * 32 + 16);
        c0 = __builtin_amdgcn_wmma_f32_16x16x32_f16(false, afr0[ks], false, b,
                                                    (short)0, c0, false, false);
        c1 = __builtin_amdgcn_wmma_f32_16x16x32_f16(false, afr1[ks], false, b,
                                                    (short)0, c1, false, false);
      }
      const float bb = B2[wrow];
#pragma unroll
      for (int r = 0; r < 8; ++r) {
        acc0[r] += (float)xs[((mb + r) << nfsh) + i] * (c0[r] + bb);
        acc1[r] += (float)xs[((16 + mb + r) << nfsh) + i] * (c1[r] + bb);
      }
    }
#pragma unroll
    for (int r = 0; r < 8; ++r) {
      atom_add_f32(&AGG[(size_t)did[mb + r] * NOUT + ot + l15], acc0[r]);
      atom_add_f32(&AGG[(size_t)did[16 + mb + r] * NOUT + ot + l15], acc1[r]);
    }
  }
}

// ===========================================================================
// Host orchestration
// ===========================================================================
extern "C" void kernel_launch(void* const* d_in, const int* in_sizes, int n_in,
                              void* d_out, int out_size, void* d_ws, size_t ws_size,
                              hipStream_t stream) {
  (void)n_in; (void)out_size; (void)ws_size;
  const int nN = in_sizes[0] / 16;   // 4096
  const int nE = in_sizes[2] / 10;   // 32768
  const float* x_in = (const float*)d_in[0];
  const int*   src  = (const int*)d_in[1];
  const int*   dst  = src + nE;
  const float* e_in = (const float*)d_in[2];
  auto Pp = [&](int i) { return (const float*)d_in[i]; };
  // param leaf indices (jax pytree order: dict keys sorted, lists in order)
  // 4:bn_edge.b 5:bn_edge.g 6:bn_node.b 7:bn_node.g
  // 8..15: edge_pred W/b x4 ; 16+20*i: mp[i] ; 76..83: node_pred W/b x4

  uint8_t* wsb = (uint8_t*)d_ws;
  size_t off = 0;
  auto alloc  = [&](size_t bytes) -> void* {
    off = (off + 255) & ~(size_t)255;
    void* p = wsb + off;
    off += bytes;
    return p;
  };
  auto allocH = [&](size_t n) { return (_Float16*)alloc(n * sizeof(_Float16)); };
  auto allocF = [&](size_t n) { return (float*)alloc(n * sizeof(float)); };

  // activation arenas (f16, K-padded strides)
  _Float16* XA  = allocH((size_t)nN * 128);
  _Float16* XB  = allocH((size_t)nN * 128);
  _Float16* EA  = allocH((size_t)nE * 128);
  _Float16* EB  = allocH((size_t)nE * 128);
  _Float16* HC  = allocH((size_t)nE * 384);
  _Float16* H1  = allocH((size_t)nE * 384);
  _Float16* H2  = allocH((size_t)nE * 128);
  _Float16* WNb = allocH((size_t)nE * 256);
  _Float16* G1  = allocH((size_t)nE * 64);
  _Float16* G2  = allocH((size_t)nE * 32);
  _Float16* G3  = allocH((size_t)nE * 32);
  _Float16* NG1 = allocH((size_t)nN * 64);
  _Float16* NG2 = allocH((size_t)nN * 32);
  _Float16* NG3 = allocH((size_t)nN * 32);
  float* AGG   = allocF((size_t)nN * 128);
  float* dmean = allocF(512);
  float* drstd = allocF(512);

  auto cdiv = [](int a, int b) { return (a + b - 1) / b; };
  auto packW = [&](const float* W, int N, int K, int Kp) -> _Float16* {
    _Float16* o = allocH((size_t)N * Kp);
    pack_wt<<<dim3(cdiv(N * Kp, 256)), 256, 0, stream>>>(W, o, N, K, Kp);
    return o;
  };
  auto packB = [&](const float* B, int K, int N, int Kp) -> _Float16* {
    _Float16* o = allocH((size_t)Kp * N);
    pack_bk<<<dim3(cdiv(Kp * N, 256)), 256, 0, stream>>>(B, o, K, N, Kp);
    return o;
  };
  auto gemm = [&](const _Float16* A, int lda, const _Float16* B, int ldb, int bt,
                  const float* bias, const _Float16* addh, int ldah,
                  const float* addf, int ldaf,
                  _Float16* Ch, int ldch, float* Cf, int ldcf,
                  int M, int N, int Npad, int K, int lre) {
    int nps = (Npad + 31) & ~31;
    wmma_gemm<<<dim3(nps / 32, M / 16), 32, 0, stream>>>(
        A, lda, B, ldb, bt, bias, addh, ldah, addf, ldaf, Ch, ldch, Cf, ldcf,
        N, K, lre);
  };
  auto bn = [&](_Float16* Xp, int M, int S, int D, const float* g, const float* b) {
    bn_stats_h<<<dim3(D), 256, 0, stream>>>(Xp, M, S, dmean, drstd);
    bn_apply_h<<<dim3(cdiv(M * D, 256)), 256, 0, stream>>>(Xp, M, S, D, dmean,
                                                           drstd, g, b);
  };

  // ---- input batchnorms (f32 stats, emit padded f16) ----
  bn_stats_f<<<dim3(16), 256, 0, stream>>>(x_in, nN, 16, dmean, drstd);
  bn_cvt<<<dim3(cdiv(nN * 32, 256)), 256, 0, stream>>>(x_in, nN, 16, XA, 32,
                                                       dmean, drstd, Pp(7), Pp(6));
  bn_stats_f<<<dim3(10), 256, 0, stream>>>(e_in, nE, 10, dmean, drstd);
  bn_cvt<<<dim3(cdiv(nE * 32, 256)), 256, 0, stream>>>(e_in, nE, 10, EA, 32,
                                                       dmean, drstd, Pp(5), Pp(4));

  struct LC { int nf, Sx, ef, Se, Sc, hid, Sh, eout, nout; };
  const LC L[3] = {
      {16, 32, 10, 32, 64, 84, 96, 32, 32},
      {32, 32, 32, 32, 96, 192, 192, 64, 64},
      {64, 64, 64, 64, 192, 384, 384, 128, 128},
  };
  const int nfsh[3] = {4, 5, 6};

  _Float16 *xc = XA, *xn = XB, *ec = EA, *en = EB;
  for (int i = 0; i < 3; ++i) {
    const LC& c = L[i];
    const int base = 16 + 20 * i;
    const float* bias_r = Pp(base + 0);
    const float *bo_b = Pp(base + 1), *bo_g = Pp(base + 2);
    const float *e1_b = Pp(base + 3), *e1_g = Pp(base + 4);
    const float *e2_b = Pp(base + 5), *e2_g = Pp(base + 6);
    const float *W1 = Pp(base + 7),  *b1 = Pp(base + 8);
    const float *W2e = Pp(base + 9), *b2e = Pp(base + 10);
    const float *Wsk = Pp(base + 11), *bsk = Pp(base + 12);
    const float *nb_b = Pp(base + 13), *nb_g = Pp(base + 14);
    const float *Wn1 = Pp(base + 15), *bn1 = Pp(base + 16);
    const float *Wn2 = Pp(base + 17), *b_n2 = Pp(base + 18);
    const float *root = Pp(base + 19);
    const int KW = 2 * c.eout;
    const int nin = 2 * c.nf + c.ef;

    _Float16* W1h  = packW(W1, c.hid, nin, c.Sc);
    _Float16* W2h  = packW(W2e, c.eout, c.hid, c.Sh);
    _Float16* Wsh  = packW(Wsk, c.eout, c.ef, c.Se);
    _Float16* Wn1h = packW(Wn1, KW, c.eout, c.eout);
    _Float16* Wn2h = packW(Wn2, c.nf * c.nout, KW, KW);
    _Float16* Rth  = packB(root, c.nf, c.nout, c.Sx);

    // ---- EdgeModel ----
    gather_concat<<<dim3(cdiv(nE * c.Sc, 256)), 256, 0, stream>>>(
        xc, c.Sx, c.nf, ec, c.Se, c.ef, src, dst, HC, c.Sc, nE);
    gemm(HC, c.Sc, W1h, c.Sc, 1, b1, nullptr, 0, nullptr, 0,
         H1, c.Sh, nullptr, 0, nE, c.hid, c.Sh, c.Sc, 1);           // lrelu
    bn(H1, nE, c.Sh, c.hid, e1_g, e1_b);
    gemm(H1, c.Sh, W2h, c.Sh, 1, b2e, nullptr, 0, nullptr, 0,
         H2, c.eout, nullptr, 0, nE, c.eout, c.eout, c.Sh, 1);      // lrelu
    gemm(ec, c.Se, Wsh, c.Se, 1, bsk, H2, c.eout, nullptr, 0,
         en, c.eout, nullptr, 0, nE, c.eout, c.eout, c.Se, 0);      // + skip
    bn(en, nE, c.eout, c.eout, e2_g, e2_b);

    // ---- NNConv ----
    gemm(en, c.eout, Wn1h, c.eout, 1, bn1, nullptr, 0, nullptr, 0,
         WNb, KW, nullptr, 0, nE, KW, KW, c.eout, 1);               // lrelu
    bn(WNb, nE, KW, KW, nb_g, nb_b);
    hipMemsetAsync(AGG, 0, (size_t)nN * c.nout * sizeof(float), stream);
    {
      dim3 g(nE / 32);
      if (KW == 64)
        nnconv_fused<2><<<g, 32, 0, stream>>>(WNb, Wn2h, b_n2, xc, c.Sx, c.nf,
                                              nfsh[i], src, dst, AGG, c.nout);
      else if (KW == 128)
        nnconv_fused<4><<<g, 32, 0, stream>>>(WNb, Wn2h, b_n2, xc, c.Sx, c.nf,
                                              nfsh[i], src, dst, AGG, c.nout);
      else
        nnconv_fused<8><<<g, 32, 0, stream>>>(WNb, Wn2h, b_n2, xc, c.Sx, c.nf,
                                              nfsh[i], src, dst, AGG, c.nout);
    }
    gemm(xc, c.Sx, Rth, c.nout, 0, bias_r, nullptr, 0, AGG, c.nout,
         xn, c.nout, nullptr, 0, nN, c.nout, c.nout, c.Sx, 0);      // +agg+bias
    bn(xn, nN, c.nout, c.nout, bo_g, bo_b);

    _Float16* t = xc; xc = xn; xn = t;
    t = ec; ec = en; en = t;
  }

  // ---- predictor ----
  gather_concat<<<dim3(cdiv(nE * 384, 256)), 256, 0, stream>>>(
      xc, 128, 128, ec, 128, 128, src, dst, HC, 384, nE);
  _Float16* E0 = packW(Pp(8), 64, 384, 384);
  _Float16* E1 = packW(Pp(10), 16, 64, 64);
  _Float16* E2 = packW(Pp(12), 8, 16, 32);
  _Float16* E3 = packW(Pp(14), 2, 8, 32);
  float* out_node = (float*)d_out;
  float* out_edge = out_node + (size_t)nN * 2;
  gemm(HC, 384, E0, 384, 1, Pp(9), nullptr, 0, nullptr, 0,
       G1, 64, nullptr, 0, nE, 64, 64, 384, 1);
  gemm(G1, 64, E1, 64, 1, Pp(11), nullptr, 0, nullptr, 0,
       G2, 32, nullptr, 0, nE, 16, 32, 64, 1);
  gemm(G2, 32, E2, 32, 1, Pp(13), nullptr, 0, nullptr, 0,
       G3, 32, nullptr, 0, nE, 8, 32, 32, 1);
  gemm(G3, 32, E3, 32, 1, Pp(15), nullptr, 0, nullptr, 0,
       nullptr, 0, out_edge, 2, nE, 2, 16, 32, 0);

  _Float16* N0 = packW(Pp(76), 64, 128, 128);
  _Float16* N1 = packW(Pp(78), 16, 64, 64);
  _Float16* N2 = packW(Pp(80), 8, 16, 32);
  _Float16* N3 = packW(Pp(82), 2, 8, 32);
  gemm(xc, 128, N0, 128, 1, Pp(77), nullptr, 0, nullptr, 0,
       NG1, 64, nullptr, 0, nN, 64, 64, 128, 1);
  gemm(NG1, 64, N1, 64, 1, Pp(79), nullptr, 0, nullptr, 0,
       NG2, 32, nullptr, 0, nN, 16, 32, 64, 1);
  gemm(NG2, 32, N2, 32, 1, Pp(81), nullptr, 0, nullptr, 0,
       NG3, 32, nullptr, 0, nN, 8, 32, 32, 1);
  gemm(NG3, 32, N3, 32, 1, Pp(83), nullptr, 0, nullptr, 0,
       nullptr, 0, out_node, 2, nN, 2, 16, 32, 0);
}
